// Topkpool_57071525429589
// MI455X (gfx1250) — matrix operand order repaired and checked by reading, not compiled, not run on hardware
//
#include <hip/hip_runtime.h>
#include <hip/hip_bf16.h>
#include <math.h>

typedef __bf16 v16bf __attribute__((ext_vector_type(16)));
typedef float  v8f   __attribute__((ext_vector_type(8)));

#define CDIV(a,b) (((a)+(b)-1)/(b))

// ---------------- utility kernels ----------------
__global__ void zero_f_kernel(float* p, long n) {
    long i = (long)blockIdx.x * blockDim.x + threadIdx.x;
    if (i < n) p[i] = 0.0f;
}
__global__ void zero_u_kernel(unsigned* p, int n) {
    int i = blockIdx.x * blockDim.x + threadIdx.x;
    if (i < n) p[i] = 0u;
}
__global__ void edge_init_kernel(const int* __restrict__ ei, int* esrc, int* edst, int E) {
    int e = blockIdx.x * blockDim.x + threadIdx.x;
    if (e < E) { esrc[e] = ei[e]; edst[e] = ei[E + e]; }
}

// ---------------- scatter-add: agg[dst] += xin[src], one wave per edge ----------------
__global__ void scatter_add_kernel(const float* __restrict__ xin, const int* __restrict__ esrc,
                                   const int* __restrict__ edst, float* __restrict__ agg, int E) {
    int wid  = (blockIdx.x * blockDim.x + threadIdx.x) >> 5;
    int lane = threadIdx.x & 31;
    if (wid >= E) return;
    int s = esrc[wid];
    if (s < 0) return;                       // masked edge
    int d = edst[wid];
    float4 v = *(const float4*)(xin + (size_t)s * 128 + lane * 4);
    float* dp = agg + (size_t)d * 128 + lane * 4;
    __hip_atomic_fetch_add(dp + 0, v.x, __ATOMIC_RELAXED, __HIP_MEMORY_SCOPE_AGENT);
    __hip_atomic_fetch_add(dp + 1, v.y, __ATOMIC_RELAXED, __HIP_MEMORY_SCOPE_AGENT);
    __hip_atomic_fetch_add(dp + 2, v.z, __ATOMIC_RELAXED, __HIP_MEMORY_SCOPE_AGENT);
    __hip_atomic_fetch_add(dp + 3, v.w, __ATOMIC_RELAXED, __HIP_MEMORY_SCOPE_AGENT);
}

// ---------------- fused dual-GEMM + bias + relu via bf16 WMMA ----------------
// h[m][n] = relu( sum_k agg[m][k]*Wr[n][k] + sum_k xin[m][k]*Wroot[n][k] + br[n] )
// A frag (16x32 bf16, ISA 7.12.2): lane row = m0+(l&15); grp = l>>4;
//   vgpr j<4: K = kb + grp*8 + 2j,2j+1 ; vgpr j>=4: K = kb + 16 + grp*8 + 2(j-4),+1
__device__ __forceinline__ v16bf load_a_frag(const float* __restrict__ A, int row, int kb, int grp) {
    v16bf a;
    const float* p = A + (size_t)row * 128 + kb;
#pragma unroll
    for (int j = 0; j < 8; ++j) {
        int koff = (j < 4) ? (grp * 8 + 2 * j) : (16 + grp * 8 + 2 * (j - 4));
        a[2 * j]     = (__bf16)p[koff];
        a[2 * j + 1] = (__bf16)p[koff + 1];
    }
    return a;
}
// B frag (32x16 bf16): B[k][n] = W[n][k]; lane col = n0+(l&15);
//   vgpr j<4: K = kb + grp*8 + 2j,2j+1 ; vgpr j>=4: K = kb + 16 + grp*8 + 2(j-4),+1
__device__ __forceinline__ v16bf load_b_frag(const float* __restrict__ W, int col, int kb, int grp) {
    v16bf b;
    const float* p = W + (size_t)col * 128 + kb;
#pragma unroll
    for (int j = 0; j < 8; ++j) {
        int koff = (j < 4) ? (grp * 8 + 2 * j) : (16 + grp * 8 + 2 * (j - 4));
        b[2 * j]     = (__bf16)p[koff];
        b[2 * j + 1] = (__bf16)p[koff + 1];
    }
    return b;
}

__global__ void gconv_gemm_kernel(const float* __restrict__ agg, const float* __restrict__ xin,
                                  const float* __restrict__ Wr, const float* __restrict__ Wroot,
                                  const float* __restrict__ br, float* __restrict__ h, int M) {
    int wave = threadIdx.x >> 5;            // 0..7 -> n tile
    int lane = threadIdx.x & 31;
    int grp  = lane >> 4;
    int m0   = blockIdx.x * 16;             // M is a multiple of 16 for all layers
    int n0   = wave * 16;
    int row  = m0 + (lane & 15);
    int col  = n0 + (lane & 15);
    v8f c = {};
#pragma unroll
    for (int kb = 0; kb < 128; kb += 32) {
        v16bf a1 = load_a_frag(agg, row, kb, grp);
        v16bf b1 = load_b_frag(Wr, col, kb, grp);
        c = __builtin_amdgcn_wmma_f32_16x16x32_bf16(false, a1, false, b1, (short)0, c, false, false);
        v16bf a2 = load_a_frag(xin, row, kb, grp);
        v16bf b2 = load_b_frag(Wroot, col, kb, grp);
        c = __builtin_amdgcn_wmma_f32_16x16x32_bf16(false, a2, false, b2, (short)0, c, false, false);
    }
    float bias = br[col];
#pragma unroll
    for (int r = 0; r < 8; ++r) {           // C/D layout: vgpr r, lane -> row m0+r+8*grp, col
        int m = m0 + r + grp * 8;
        h[(size_t)m * 128 + col] = fmaxf(c[r] + bias, 0.0f);
    }
}

// ---------------- topk scoring ----------------
__global__ void wnorm_kernel(const float* __restrict__ w, float* out) {
    __shared__ float s[128];
    int t = threadIdx.x;
    float v = w[t];
    s[t] = v * v;
    __syncthreads();
    for (int off = 64; off > 0; off >>= 1) { if (t < off) s[t] += s[t + off]; __syncthreads(); }
    if (t == 0) *out = sqrtf(s[0]);
}

__device__ __forceinline__ unsigned f2ord(float f) {      // monotone float->uint
    unsigned b = __float_as_uint(f);
    return (b & 0x80000000u) ? ~b : (b | 0x80000000u);
}
__device__ __forceinline__ float ord2f(unsigned u) {
    unsigned b = (u & 0x80000000u) ? (u & 0x7FFFFFFFu) : ~u;
    return __uint_as_float(b);
}

__global__ void score_kernel(const float* __restrict__ h, const float* __restrict__ w,
                             const float* __restrict__ wnorm, float* score, unsigned* key, int M) {
    int node = (blockIdx.x * blockDim.x + threadIdx.x) >> 5;
    int lane = threadIdx.x & 31;
    if (node >= M) return;
    float4 hv = *(const float4*)(h + (size_t)node * 128 + lane * 4);
    float4 wv = *(const float4*)(w + lane * 4);
    float p = hv.x * wv.x + hv.y * wv.y + hv.z * wv.z + hv.w * wv.w;
    for (int off = 16; off > 0; off >>= 1) p += __shfl_xor(p, off, 32);
    if (lane == 0) {
        float sc = tanhf(p / *wnorm);
        score[node] = sc;
        key[node] = f2ord(sc);
    }
}

// ---------------- 2-pass radix select of k-th largest key ----------------
__global__ void hist_hi_kernel(const unsigned* __restrict__ key, unsigned* hist, int M) {
    int i = blockIdx.x * blockDim.x + threadIdx.x;
    if (i < M) atomicAdd(&hist[key[i] >> 16], 1u);
}
__global__ void scan_hi_kernel(const unsigned* __restrict__ hist, unsigned* sel, int k) {
    unsigned cum = 0;
    for (int b = 65535; b >= 0; --b) {
        unsigned c = hist[b];
        if (cum + c >= (unsigned)k) { sel[0] = (unsigned)b; sel[1] = cum; return; }
        cum += c;
    }
    sel[0] = 0u; sel[1] = cum;
}
__global__ void hist_lo_kernel(const unsigned* __restrict__ key, unsigned* hist,
                               const unsigned* __restrict__ sel, int M) {
    int i = blockIdx.x * blockDim.x + threadIdx.x;
    if (i < M) {
        unsigned kk = key[i];
        if ((kk >> 16) == sel[0]) atomicAdd(&hist[kk & 0xFFFFu], 1u);
    }
}
__global__ void scan_lo_kernel(const unsigned* __restrict__ hist, unsigned* sel, int k) {
    unsigned rem = (unsigned)k - sel[1];
    unsigned cum = 0;
    for (int b = 65535; b >= 0; --b) {
        unsigned c = hist[b];
        if (cum + c >= rem) {
            sel[2] = (sel[0] << 16) | (unsigned)b;   // threshold key T
            sel[3] = rem - cum;                      // ties to keep (k - count(>T))
            sel[4] = 0u; sel[5] = 0u;                // reset claim counters
            return;
        }
        cum += c;
    }
}

// ---------------- compaction: xp[slot] = h[node] * score[node] ----------------
__global__ void compact_kernel(const float* __restrict__ h, const float* __restrict__ score,
                               const unsigned* __restrict__ key, unsigned* sel,
                               int* __restrict__ newidx, float* __restrict__ xp, int M, int k) {
    int node = (blockIdx.x * blockDim.x + threadIdx.x) >> 5;
    int lane = threadIdx.x & 31;
    if (node >= M) return;
    unsigned T = sel[2], tn = sel[3];
    int slot = -1;
    if (lane == 0) {
        unsigned kk = key[node];
        if (kk > T) slot = (int)atomicAdd(&sel[4], 1u);                 // [0, k-tn)
        else if (kk == T) {
            unsigned j = atomicAdd(&sel[5], 1u);
            if (j < tn) slot = (int)((unsigned)k - tn + j);             // [k-tn, k)
        }
        newidx[node] = slot;
    }
    slot = __shfl(slot, 0, 32);
    if (slot < 0) return;
    float sc = score[node];
    float4 v = *(const float4*)(h + (size_t)node * 128 + lane * 4);
    v.x *= sc; v.y *= sc; v.z *= sc; v.w *= sc;
    *(float4*)(xp + (size_t)slot * 128 + lane * 4) = v;
}

__global__ void remap_edges_kernel(int* esrc, int* edst, const int* __restrict__ newidx, int E) {
    int e = blockIdx.x * blockDim.x + threadIdx.x;
    if (e >= E) return;
    int s = esrc[e];
    if (s < 0) return;
    int ns = newidx[s], nd = newidx[edst[e]];
    if (ns < 0 || nd < 0) esrc[e] = -1;
    else { esrc[e] = ns; edst[e] = nd; }
}

// ---------------- readout: max || mean ----------------
__global__ void readout_init_kernel(unsigned* rmaxk, float* rmean) {
    int i = threadIdx.x;                    // 128 threads
    rmaxk[i] = 0u;                          // smallest ordered key
    rmean[i] = 0.0f;
}
__global__ void readout_accum_kernel(const float* __restrict__ xp, unsigned* rmaxk,
                                     float* rmean, int M) {
    int node = (blockIdx.x * blockDim.x + threadIdx.x) >> 5;
    int lane = threadIdx.x & 31;
    if (node >= M) return;
    float4 v = *(const float4*)(xp + (size_t)node * 128 + lane * 4);
    float vals[4] = {v.x, v.y, v.z, v.w};
#pragma unroll
    for (int j = 0; j < 4; ++j) {
        int f = lane * 4 + j;
        atomicMax(&rmaxk[f], f2ord(vals[j]));
        __hip_atomic_fetch_add(&rmean[f], vals[j], __ATOMIC_RELAXED, __HIP_MEMORY_SCOPE_AGENT);
    }
}
__global__ void readout_final_kernel(const unsigned* __restrict__ rmaxk,
                                     const float* __restrict__ rmean, float* rsum, float invK) {
    int f = threadIdx.x;                    // 128 threads
    rsum[f]       += ord2f(rmaxk[f]);
    rsum[128 + f] += rmean[f] * invK;
}

// ---------------- final MLP on [1,256] ----------------
__global__ void mlp_kernel(const float* __restrict__ r, const float* __restrict__ W1,
                           const float* __restrict__ b1, const float* __restrict__ W2,
                           const float* __restrict__ b2, const float* __restrict__ W3,
                           const float* __restrict__ b3, float* __restrict__ out) {
    __shared__ float z1[128], z2[64];
    int t = threadIdx.x;                    // 256 threads
    if (t < 128) {
        float acc = b1[t];
        for (int i = 0; i < 256; ++i) acc += r[i] * W1[t * 256 + i];
        z1[t] = fmaxf(acc, 0.0f);
    }
    __syncthreads();
    if (t < 64) {
        float acc = b2[t];
        for (int i = 0; i < 128; ++i) acc += z1[i] * W2[t * 128 + i];
        z2[t] = fmaxf(acc, 0.0f);
    }
    __syncthreads();
    if (t < 2) {
        float acc = b3[t];
        for (int i = 0; i < 64; ++i) acc += z2[i] * W3[t * 64 + i];
        out[t] = acc;
    }
}

// ---------------- host orchestration ----------------
extern "C" void kernel_launch(void* const* d_in, const int* in_sizes, int n_in,
                              void* d_out, int out_size, void* d_ws, size_t ws_size,
                              hipStream_t stream) {
    const float* x      = (const float*)d_in[0];
    const int*   ei     = (const int*)d_in[1];
    const float* Wr[3]    = {(const float*)d_in[2],  (const float*)d_in[6],  (const float*)d_in[10]};
    const float* br[3]    = {(const float*)d_in[3],  (const float*)d_in[7],  (const float*)d_in[11]};
    const float* Wroot[3] = {(const float*)d_in[4],  (const float*)d_in[8],  (const float*)d_in[12]};
    const float* pw[3]    = {(const float*)d_in[5],  (const float*)d_in[9],  (const float*)d_in[13]};
    const float* Wl1 = (const float*)d_in[14]; const float* bl1 = (const float*)d_in[15];
    const float* Wl2 = (const float*)d_in[16]; const float* bl2 = (const float*)d_in[17];
    const float* Wl3 = (const float*)d_in[18]; const float* bl3 = (const float*)d_in[19];

    const int N = in_sizes[0] / 128;
    const int E = in_sizes[1] / 2;

    // workspace carve-out (~160 MB for N=100000, E=600000)
    float*    G      = (float*)d_ws;                 // agg / rotating node-feature buffer
    float*    H      = G + (size_t)N * 128;          // conv output
    float*    P      = H + (size_t)N * 128;          // pooled features
    float*    score  = P + (size_t)N * 128;          // [N]
    unsigned* key    = (unsigned*)(score + N);       // [N]
    int*      newidx = (int*)(key + N);              // [N]
    int*      esrc   = newidx + N;                   // [E]
    int*      edst   = esrc + E;                     // [E]
    unsigned* hist   = (unsigned*)(edst + E);        // [65536]
    unsigned* sel    = hist + 65536;                 // [8]: selHi,cAbove,thrKey,tieNeeded,ctrGt,ctrEq
    float*    wnorm  = (float*)(sel + 8);            // [1] (+pad)
    unsigned* rmaxk  = (unsigned*)(wnorm + 4);       // [128]
    float*    rmean  = (float*)(rmaxk + 128);        // [128]
    float*    rsum   = rmean + 128;                  // [256]

    // init: readout accumulator + mutable edge copy (inputs stay untouched)
    zero_f_kernel<<<1, 256, 0, stream>>>(rsum, 256);
    edge_init_kernel<<<CDIV(E, 256), 256, 0, stream>>>(ei, esrc, edst, E);

    auto layer = [&](const float* xin, int M, int k, int li,
                     float* agg, float* h, float* xp, bool remap) {
        zero_f_kernel<<<CDIV((long)M * 128, 256), 256, 0, stream>>>(agg, (long)M * 128);
        scatter_add_kernel<<<CDIV(E * 32, 256), 256, 0, stream>>>(xin, esrc, edst, agg, E);
        gconv_gemm_kernel<<<M / 16, 256, 0, stream>>>(agg, xin, Wr[li], Wroot[li], br[li], h, M);
        wnorm_kernel<<<1, 128, 0, stream>>>(pw[li], wnorm);
        score_kernel<<<CDIV(M * 32, 256), 256, 0, stream>>>(h, pw[li], wnorm, score, key, M);
        zero_u_kernel<<<CDIV(65536, 256), 256, 0, stream>>>(hist, 65536);
        hist_hi_kernel<<<CDIV(M, 256), 256, 0, stream>>>(key, hist, M);
        scan_hi_kernel<<<1, 1, 0, stream>>>(hist, sel, k);
        zero_u_kernel<<<CDIV(65536, 256), 256, 0, stream>>>(hist, 65536);
        hist_lo_kernel<<<CDIV(M, 256), 256, 0, stream>>>(key, hist, sel, M);
        scan_lo_kernel<<<1, 1, 0, stream>>>(hist, sel, k);
        compact_kernel<<<CDIV(M * 32, 256), 256, 0, stream>>>(h, score, key, sel, newidx, xp, M, k);
        if (remap) remap_edges_kernel<<<CDIV(E, 256), 256, 0, stream>>>(esrc, edst, newidx, E);
        readout_init_kernel<<<1, 128, 0, stream>>>(rmaxk, rmean);
        readout_accum_kernel<<<CDIV(k * 32, 256), 256, 0, stream>>>(xp, rmaxk, rmean, k);
        readout_final_kernel<<<1, 128, 0, stream>>>(rmaxk, rmean, rsum, 1.0f / (float)k);
    };

    const int k1 = (4 * N + 4) / 5;          // ceil(0.8*N)  -> 80000
    const int k2 = (4 * k1 + 4) / 5;         //              -> 64000
    const int k3 = (4 * k2 + 4) / 5;         //              -> 51200

    // buffer rotation: inputs of each layer stay live through its GEMM only
    layer(x, N,  k1, 0, G, H, P, true);      // xin=d_in, agg=G, h=H, pooled->P
    layer(P, k1, k2, 1, G, H, G, true);      // agg=G (dead after gemm) reused as pooled
    layer(G, k2, k3, 2, P, H, P, false);     // agg=P reused as pooled; no remap needed

    mlp_kernel<<<1, 256, 0, stream>>>(rsum, Wl1, bl1, Wl2, bl2, Wl3, bl3, (float*)d_out);
}
